// ST_ECGFormer_88948772700628
// MI455X (gfx1250) — matrix operand order repaired and checked by reading, not compiled, not run on hardware
//
#include <hip/hip_runtime.h>
#include <hip/hip_bf16.h>
#include <math.h>

// ---- model dims (compile-time) ----
#define SEQ_LEN 252
#define WIN     96
#define LEADS   12
#define S_TOK   264            // SEQ_LEN + LEADS
#define E_DIM   128
#define HEADS   8
#define HD      16             // E/HEADS
#define FF_DIM  512
#define LAYERS  12
#define BATCH   128
#define BS      (BATCH * S_TOK)   // 33792, multiple of 64
#define SP      288               // padded key length: 18*16 = 9*32
#define QT      17                // ceil(264/16)

typedef _Float16 v16h __attribute__((ext_vector_type(16)));
typedef _Float16 v8h  __attribute__((ext_vector_type(8)));
typedef float    v8f  __attribute__((ext_vector_type(8)));
typedef int      v4i  __attribute__((ext_vector_type(4)));

typedef __attribute__((address_space(1))) v4i gv4i;   // global int4
typedef __attribute__((address_space(3))) v4i lv4i;   // LDS int4

#if defined(__has_builtin)
#  if __has_builtin(__builtin_amdgcn_global_load_async_to_lds_b128)
#    define HAS_ASYNC_LDS 1
#  else
#    define HAS_ASYNC_LDS 0
#  endif
#  if __has_builtin(__builtin_amdgcn_s_wait_asynccnt)
#    define HAS_WAIT_ASYNC 1
#  else
#    define HAS_WAIT_ASYNC 0
#  endif
#else
#  define HAS_ASYNC_LDS 0
#  define HAS_WAIT_ASYNC 0
#endif

// build 16-half fragment from two aligned 8-half runs
static __device__ inline v16h cat8(v8h lo, v8h hi) {
  return __builtin_shufflevector(lo, hi, 0, 1, 2, 3, 4, 5, 6, 7,
                                 8, 9, 10, 11, 12, 13, 14, 15);
}

// ============================================================
// fp32 -> f16 elementwise cast (weights, once per launch)
// ============================================================
__global__ void cast_f16_kernel(const float* __restrict__ src,
                                _Float16* __restrict__ dst, int n) {
  int i = blockIdx.x * 256 + threadIdx.x;
  if (i < n) dst[i] = (_Float16)src[i];
}

// ============================================================
// Embedding: conv1d(k=1) + cls + pos + table gathers
// writes fp32 master h and f16 mirror h16
// ============================================================
__global__ void embed_kernel(const float* __restrict__ x,
                             const int* __restrict__ t_idx,
                             const int* __restrict__ s_idx,
                             const float* __restrict__ conv_w,
                             const float* __restrict__ conv_b,
                             const float* __restrict__ pos,
                             const float* __restrict__ time_tab,
                             const float* __restrict__ spat_tab,
                             const float* __restrict__ cls_tok,
                             float* __restrict__ h,
                             _Float16* __restrict__ h16) {
  int row = blockIdx.x;          // b*S_TOK + s
  int b = row / S_TOK;
  int s = row - b * S_TOK;
  int e = threadIdx.x;           // 0..127
  float val;
  if (s < LEADS) {
    val = cls_tok[s * E_DIM + e] + pos[s * E_DIM + e];
  } else {
    int t = s - LEADS;
    const float* xr = x + ((size_t)b * SEQ_LEN + t) * WIN;
    const float* wr = conv_w + (size_t)e * WIN;
    float acc = 0.f;
    #pragma unroll 8
    for (int w = 0; w < WIN; ++w) acc = fmaf(xr[w], wr[w], acc);
    int ti = t_idx[b * SEQ_LEN + t];
    int si = s_idx[b * SEQ_LEN + t];
    val = acc + conv_b[e] + pos[s * E_DIM + e]
        + time_tab[ti * E_DIM + e] + spat_tab[si * E_DIM + e];
  }
  h[(size_t)row * E_DIM + e] = val;
  h16[(size_t)row * E_DIM + e] = (_Float16)val;
}

// ============================================================
// WMMA GEMM:  C[M,N] = A[M,K](f16) * W[N,K](f16)^T + bias[N]
// 4 waves / block, block tile 64x64, W tile staged in LDS via
// async global->LDS; each wave: 1 A frag reused over 4 WMMAs.
// B fragments are all preloaded so the 4 WMMAs issue back-to-back.
// Optional fp32 and/or f16 outputs, optional relu.
// ============================================================
__global__ void gemm_wmma_kernel(const _Float16* __restrict__ A,
                                 const _Float16* __restrict__ W,
                                 const float* __restrict__ bias,
                                 float* __restrict__ C32,
                                 _Float16* __restrict__ C16,
                                 int M, int N, int K, int relu) {
  __shared__ __align__(16) _Float16 wtile[64][32];

  int nblocks = N >> 6;
  int nb = blockIdx.x % nblocks;
  int mb = blockIdx.x / nblocks;
  int ntBase = nb << 6;

  int tid  = threadIdx.x;        // 0..127
  int wave = tid >> 5;           // 0..3
  int lane = tid & 31;
  int hw   = lane >> 4;          // half-wave
  int l16  = lane & 15;

  int mbase = (mb << 6) + (wave << 4);       // 16 rows per wave
  const _Float16* arow = A + (size_t)(mbase + l16) * K;
  int kbA = hw << 3;                          // 0 or 8

  v8f acc[4] = {};

  for (int k0 = 0; k0 < K; k0 += 32) {
    // ---- stage 64x32 W tile into LDS (one b128 chunk x2 per thread) ----
    #pragma unroll
    for (int c = 0; c < 2; ++c) {
      int chunk = tid + c * 128;              // 0..255
      int row = chunk >> 2;                   // 0..63
      int seg = chunk & 3;                    // 0..3  (8 halfs each)
      const _Float16* gp = W + (size_t)(ntBase + row) * K + k0 + seg * 8;
      _Float16* lp = &wtile[row][seg * 8];
#if HAS_ASYNC_LDS
      __builtin_amdgcn_global_load_async_to_lds_b128(
          (gv4i*)(void*)gp, (lv4i*)(void*)lp, 0, 0);
#else
      *(v8h*)lp = *(const v8h*)gp;
#endif
    }
#if HAS_ASYNC_LDS && HAS_WAIT_ASYNC
    __builtin_amdgcn_s_wait_asynccnt(0);
#endif
    __syncthreads();

    // ---- A fragment: two contiguous 8-half runs per lane ----
    v8h alo = *(const v8h*)(arow + k0 + kbA);
    v8h ahi = *(const v8h*)(arow + k0 + 16 + kbA);
    v16h a = cat8(alo, ahi);
    __builtin_prefetch(arow + k0 + 32, 0, 0);

    // ---- preload all 4 B fragments (overlap DS latency) ----
    v16h bm[4];
    #pragma unroll
    for (int j = 0; j < 4; ++j) {
      const _Float16* bp = &wtile[(j << 4) + l16][hw << 3];
      v8h blo = *(const v8h*)(bp);
      v8h bhi = *(const v8h*)(bp + 16);
      bm[j] = cat8(blo, bhi);
    }

    // ---- 4 independent WMMAs, back-to-back issue ----
    #pragma unroll
    for (int j = 0; j < 4; ++j) {
      acc[j] = __builtin_amdgcn_wmma_f32_16x16x32_f16(false, a, false, bm[j],
                                                      (short)0, acc[j],
                                                      false, false);
    }
    __syncthreads();
  }

  // ---- epilogue ----
  #pragma unroll
  for (int j = 0; j < 4; ++j) {
    int n = ntBase + (j << 4) + l16;
    float bn = bias[n];
    #pragma unroll
    for (int r = 0; r < 8; ++r) {
      float v = acc[j][r] + bn;
      if (relu) v = fmaxf(v, 0.f);
      size_t idx = (size_t)(mbase + r + (hw << 3)) * N + n;
      if (C32) C32[idx] = v;
      if (C16) C16[idx] = (_Float16)v;
    }
  }
}

// ============================================================
// Attention: one wave per (b, head, 16-query tile).
// qkv (f16): [BS, 384] (Q|K|V).  ctx16 (f16): [BS, 128]
// ============================================================
__global__ void attn_kernel(const _Float16* __restrict__ qkv,
                            _Float16* __restrict__ ctx16) {
  __shared__ __align__(16) float    sc[16][SP];   // fp32 score strip
  __shared__ __align__(16) _Float16 pr[16][SP];   // f16 probabilities

  int bi = blockIdx.x;
  int qt = bi % QT; bi /= QT;
  int hh = bi % HEADS;
  int b  = bi / HEADS;

  int lane = threadIdx.x;
  int hw   = lane >> 4;
  int l16  = lane & 15;
  int kbA  = hw << 3;

  // ---- Q fragment (16x32, D=16 zero-padded), scaled 1/sqrt(16) ----
  int sq0 = (qt << 4) + l16;
  int sqc = sq0 < S_TOK ? sq0 : S_TOK - 1;
  const _Float16* qrow = qkv + ((size_t)(b * S_TOK + sqc)) * 384 + hh * HD;
  v8h zro = {};
  v8h qlo = *(const v8h*)(qrow + kbA);
  qlo = qlo * (_Float16)0.25f;
  v16h aq = cat8(qlo, zro);          // upper 16 K-slots are zero padding

  // ---- scores: 18 key tiles of 16 ----
  for (int jt = 0; jt < SP / 16; ++jt) {
    int sk0 = (jt << 4) + l16;
    int skc = sk0 < S_TOK ? sk0 : S_TOK - 1;
    v16h bk;
    {
      v8h blo = zro, bhi = zro;
      {                              // inner dim d = 8*hw + s, s in [0,8)
        const _Float16* krow =
            qkv + ((size_t)(b * S_TOK + skc)) * 384 + 128 + hh * HD;
        blo = *(const v8h*)(krow + kbA);
      }
      bk = cat8(blo, bhi);
    }
    v8f cs = {};
    cs = __builtin_amdgcn_wmma_f32_16x16x32_f16(false, aq, false, bk,
                                                (short)0, cs, false, false);
    int col = (jt << 4) + l16;
    bool valid = col < S_TOK;
    #pragma unroll
    for (int r = 0; r < 8; ++r)
      sc[r + (hw << 3)][col] = valid ? cs[r] : -1e30f;
  }
  __syncthreads();

  // ---- softmax per row (lanes 0..15) ----
  if (lane < 16) {
    int mrow = lane;
    float mx = -1e30f;
    for (int c = 0; c < SP; ++c) mx = fmaxf(mx, sc[mrow][c]);
    float sum = 0.f;
    for (int c = 0; c < SP; ++c) {
      float ev = expf(sc[mrow][c] - mx);
      sc[mrow][c] = ev;
      sum += ev;
    }
    float inv = 1.f / sum;
    for (int c = 0; c < SP; ++c)
      pr[mrow][c] = (_Float16)(sc[mrow][c] * inv);
  }
  __syncthreads();

  // ---- context = P @ V : 9 K-steps of 32 keys ----
  const _Float16* vcol = qkv + 256 + hh * HD + l16;   // fixed output dim/lane
  v8f co = {};
  for (int jj = 0; jj < SP / 32; ++jj) {
    const _Float16* prow = &pr[l16][jj << 5];
    v8h plo = *(const v8h*)(prow + kbA);
    v8h phi = *(const v8h*)(prow + 16 + kbA);
    v16h ap = cat8(plo, phi);
    v16h bv;
    #pragma unroll
    for (int e = 0; e < 16; ++e) {
      int sv = (jj << 5) + ((e >> 3) << 4) + (hw << 3) + (e & 7);
      if (sv > S_TOK - 1) sv = S_TOK - 1;             // probs are 0 there
      bv[e] = vcol[(size_t)(b * S_TOK + sv) * 384];
    }
    co = __builtin_amdgcn_wmma_f32_16x16x32_f16(false, ap, false, bv,
                                                (short)0, co, false, false);
  }
  #pragma unroll
  for (int r = 0; r < 8; ++r) {
    int sq = (qt << 4) + r + (hw << 3);
    if (sq < S_TOK)
      ctx16[((size_t)(b * S_TOK + sq)) * E_DIM + hh * HD + l16] =
          (_Float16)co[r];
  }
}

// ============================================================
// h = LayerNorm(h + res); writes fp32 h and f16 h16
// ============================================================
__global__ void add_ln_kernel(float* __restrict__ h,
                              _Float16* __restrict__ h16,
                              const float* __restrict__ res,
                              const float* __restrict__ g,
                              const float* __restrict__ bta) {
  __shared__ float red[E_DIM];
  int row = blockIdx.x;
  int t = threadIdx.x;
  float x = h[(size_t)row * E_DIM + t] + res[(size_t)row * E_DIM + t];
  red[t] = x;
  __syncthreads();
  for (int off = 64; off > 0; off >>= 1) {
    if (t < off) red[t] += red[t + off];
    __syncthreads();
  }
  float mean = red[0] * (1.f / E_DIM);
  __syncthreads();
  float d = x - mean;
  red[t] = d * d;
  __syncthreads();
  for (int off = 64; off > 0; off >>= 1) {
    if (t < off) red[t] += red[t + off];
    __syncthreads();
  }
  float var = red[0] * (1.f / E_DIM);
  float y = d * rsqrtf(var + 1e-5f) * g[t] + bta[t];
  h[(size_t)row * E_DIM + t] = y;
  h16[(size_t)row * E_DIM + t] = (_Float16)y;
}

// ============================================================
// Host-side launch sequence
// ============================================================
extern "C" void kernel_launch(void* const* d_in, const int* in_sizes, int n_in,
                              void* d_out, int out_size, void* d_ws, size_t ws_size,
                              hipStream_t stream) {
  (void)in_sizes; (void)n_in; (void)out_size; (void)ws_size;

  const float* x        = (const float*)d_in[0];
  const int*   t_idx    = (const int*)  d_in[1];
  const int*   s_idx    = (const int*)  d_in[2];
  const float* conv_w   = (const float*)d_in[3];
  const float* conv_b   = (const float*)d_in[4];
  const float* pos      = (const float*)d_in[5];
  const float* time_tab = (const float*)d_in[6];
  const float* spat_tab = (const float*)d_in[7];
  const float* cls_tok  = (const float*)d_in[8];
  const float* Wqkv     = (const float*)d_in[9];
  const float* bqkv     = (const float*)d_in[10];
  const float* Wo       = (const float*)d_in[11];
  const float* bo       = (const float*)d_in[12];
  const float* W1       = (const float*)d_in[13];
  const float* b1       = (const float*)d_in[14];
  const float* W2       = (const float*)d_in[15];
  const float* b2       = (const float*)d_in[16];
  const float* ln1_g    = (const float*)d_in[17];
  const float* ln1_b    = (const float*)d_in[18];
  const float* ln2_g    = (const float*)d_in[19];
  const float* ln2_b    = (const float*)d_in[20];

  // ---- workspace carve-up ----
  float* h   = (float*)d_ws;                        // [BS,128] fp32
  float* c32 = h + (size_t)BS * E_DIM;              // [BS,128] fp32 (gemm out)
  _Float16* h16   = (_Float16*)(c32 + (size_t)BS * E_DIM);   // [BS,128]
  _Float16* act16 = h16 + (size_t)BS * E_DIM;       // [BS,512] (qkv / ff1)
  _Float16* ctx16 = act16 + (size_t)BS * FF_DIM;    // [BS,128]
  _Float16* wq16  = ctx16 + (size_t)BS * E_DIM;     // 12*384*128
  _Float16* wo16  = wq16 + (size_t)LAYERS * 3 * E_DIM * E_DIM;
  _Float16* w116  = wo16 + (size_t)LAYERS * E_DIM * E_DIM;
  _Float16* w216  = w116 + (size_t)LAYERS * FF_DIM * E_DIM;

  // ---- cast all weights to f16 once ----
  {
    int nq = LAYERS * 3 * E_DIM * E_DIM;
    int no = LAYERS * E_DIM * E_DIM;
    int n1 = LAYERS * FF_DIM * E_DIM;
    int n2 = LAYERS * E_DIM * FF_DIM;
    cast_f16_kernel<<<(nq + 255) / 256, 256, 0, stream>>>(Wqkv, wq16, nq);
    cast_f16_kernel<<<(no + 255) / 256, 256, 0, stream>>>(Wo, wo16, no);
    cast_f16_kernel<<<(n1 + 255) / 256, 256, 0, stream>>>(W1, w116, n1);
    cast_f16_kernel<<<(n2 + 255) / 256, 256, 0, stream>>>(W2, w216, n2);
  }

  // ---- embedding ----
  embed_kernel<<<BS, E_DIM, 0, stream>>>(x, t_idx, s_idx, conv_w, conv_b, pos,
                                         time_tab, spat_tab, cls_tok, h, h16);

  const int mblk = BS / 64;   // 528
  for (int i = 0; i < LAYERS; ++i) {
    const _Float16* wq_i = wq16 + (size_t)i * 3 * E_DIM * E_DIM;
    const _Float16* wo_i = wo16 + (size_t)i * E_DIM * E_DIM;
    const _Float16* w1_i = w116 + (size_t)i * FF_DIM * E_DIM;
    const _Float16* w2_i = w216 + (size_t)i * E_DIM * FF_DIM;
    const float* bqkv_i = bqkv + (size_t)i * 3 * E_DIM;
    const float* bo_i   = bo   + (size_t)i * E_DIM;
    const float* b1_i   = b1   + (size_t)i * FF_DIM;
    const float* b2_i   = b2   + (size_t)i * E_DIM;

    // QKV: h16 x wq^T -> act16 [BS,384] (f16 only)
    gemm_wmma_kernel<<<mblk * (3 * E_DIM / 64), 128, 0, stream>>>(
        h16, wq_i, bqkv_i, nullptr, act16, BS, 3 * E_DIM, E_DIM, 0);

    // attention -> ctx16 [BS,128]
    attn_kernel<<<BATCH * HEADS * QT, 32, 0, stream>>>(act16, ctx16);

    // O-proj: ctx16 x wo^T -> c32 [BS,128] (fp32 for residual)
    gemm_wmma_kernel<<<mblk * (E_DIM / 64), 128, 0, stream>>>(
        ctx16, wo_i, bo_i, c32, nullptr, BS, E_DIM, E_DIM, 0);

    // h = LN(h + proj)
    add_ln_kernel<<<BS, E_DIM, 0, stream>>>(h, h16, c32, ln1_g + i * E_DIM,
                                            ln1_b + i * E_DIM);

    // FFN1 (relu): h16 x w1^T -> act16 [BS,512] (f16 only)
    gemm_wmma_kernel<<<mblk * (FF_DIM / 64), 128, 0, stream>>>(
        h16, w1_i, b1_i, nullptr, act16, BS, FF_DIM, E_DIM, 1);

    // FFN2: act16 x w2^T -> c32 [BS,128] (fp32 for residual)
    gemm_wmma_kernel<<<mblk * (E_DIM / 64), 128, 0, stream>>>(
        act16, w2_i, b2_i, c32, nullptr, BS, E_DIM, FF_DIM, 0);

    // h = LN(h + ffn)
    add_ln_kernel<<<BS, E_DIM, 0, stream>>>(h, h16, c32, ln2_g + i * E_DIM,
                                            ln2_b + i * E_DIM);
  }

  (void)hipMemcpyAsync(d_out, h, (size_t)BS * E_DIM * sizeof(float),
                       hipMemcpyDeviceToDevice, stream);
}
